// NP2MT_59751585022160
// MI455X (gfx1250) — compile-verified, matrix operand order checked by reading, not run on hardware
//
#include <hip/hip_runtime.h>
#include <hip/hip_bf16.h>
#include <math.h>

typedef __attribute__((ext_vector_type(16))) _Float16 v16h;
typedef __attribute__((ext_vector_type(8)))  _Float16 v8h;
typedef __attribute__((ext_vector_type(8)))  float    v8f;

#define SOP_TOK 2
#define EOP_TOK 3

static __device__ __forceinline__ int imin(int a, int b) { return a < b ? a : b; }

// Load a 16x32 f16 WMMA fragment chunk for this lane.
// Per CDNA5 ISA (CDNA5 ISA): lane L (row = L&15, hi = L>>4) holds
// K = {hi*8 .. hi*8+7} and K = {16+hi*8 .. 16+hi*8+7}: two contiguous 16B chunks.
static __device__ __forceinline__ v16h ldfrag(const _Float16* p) {
    v8h lo = *(const v8h*)(p);
    v8h hi = *(const v8h*)(p + 16);
    v16h r;
#pragma unroll
    for (int i = 0; i < 8; ++i) { r[i] = lo[i]; r[8 + i] = hi[i]; }
    return r;
}

// ---------------------------------------------------------------------------
// f16 WMMA GEMM: C(MxN) = A(MxK) * Bt(NxK)^T [+ bias(N)]
// Block = 128 threads (4 waves). Each wave owns 4 adjacent 16-col tiles and
// reuses a single A fragment across them (4x A-traffic reduction).
// Grid: (N/256, M/16). M mult of 16, N mult of 256, K mult of 32.
// ---------------------------------------------------------------------------
__global__ __launch_bounds__(128)
void gemm_wmma_f16(const _Float16* __restrict__ A, int lda,
                   const _Float16* __restrict__ Bt, int ldb,
                   void* __restrict__ Cv, int ldc,
                   const float* __restrict__ bias,
                   int K, int outF16)
{
    int lane = threadIdx.x & 31;
    int wave = threadIdx.x >> 5;
    int n0 = (blockIdx.x * 4 + wave) * 64;      // wave covers cols [n0, n0+64)
    int m0 = blockIdx.y * 16;
    int hi = (lane >> 4) * 8;

    const _Float16* Arow = A + (size_t)(m0 + (lane & 15)) * lda;
    const _Float16* Brow0 = Bt + (size_t)(n0      + (lane & 15)) * ldb;
    const _Float16* Brow1 = Bt + (size_t)(n0 + 16 + (lane & 15)) * ldb;
    const _Float16* Brow2 = Bt + (size_t)(n0 + 32 + (lane & 15)) * ldb;
    const _Float16* Brow3 = Bt + (size_t)(n0 + 48 + (lane & 15)) * ldb;

    v8f acc0 = {}, acc1 = {}, acc2 = {}, acc3 = {};
    for (int k0 = 0; k0 < K; k0 += 32) {
        v16h a  = ldfrag(Arow + k0 + hi);
        v16h b0 = ldfrag(Brow0 + k0 + hi);
        acc0 = __builtin_amdgcn_wmma_f32_16x16x32_f16(false, a, false, b0, (short)0, acc0, false, false);
        v16h b1 = ldfrag(Brow1 + k0 + hi);
        acc1 = __builtin_amdgcn_wmma_f32_16x16x32_f16(false, a, false, b1, (short)0, acc1, false, false);
        v16h b2 = ldfrag(Brow2 + k0 + hi);
        acc2 = __builtin_amdgcn_wmma_f32_16x16x32_f16(false, a, false, b2, (short)0, acc2, false, false);
        v16h b3 = ldfrag(Brow3 + k0 + hi);
        acc3 = __builtin_amdgcn_wmma_f32_16x16x32_f16(false, a, false, b3, (short)0, acc3, false, false);
    }

    // C layout: lane L -> col = L&15; rows (L>>4)*8 + r  (r = 0..7)
    int col = lane & 15;
    int rb  = (lane >> 4) * 8;
    v8f accs[4] = {acc0, acc1, acc2, acc3};
#pragma unroll
    for (int tt = 0; tt < 4; ++tt) {
        int nc = n0 + tt * 16 + col;
        float bv = bias ? bias[nc] : 0.f;
        if (outF16) {
            _Float16* C = (_Float16*)Cv;
#pragma unroll
            for (int r = 0; r < 8; ++r)
                C[(size_t)(m0 + rb + r) * ldc + nc] = (_Float16)(accs[tt][r] + bv);
        } else {
            float* C = (float*)Cv;
#pragma unroll
            for (int r = 0; r < 8; ++r)
                C[(size_t)(m0 + rb + r) * ldc + nc] = accs[tt][r] + bv;
        }
    }
}

// ---------------------------------------------------------------------------
// Tiled convert+transpose: fp32 W(K,N) -> f16 Wt(N,K).
// 32x32 LDS tile (+1 pad vs 64 banks); both global sides coalesced.
// Grid: (N/32, K/32), block (32,8).
// ---------------------------------------------------------------------------
__global__ __launch_bounds__(256)
void cvt_transpose(const float* __restrict__ W, _Float16* __restrict__ Wt,
                   int K, int N)
{
    __shared__ _Float16 tile[32][33];
    int n0 = blockIdx.x * 32;
    int k0 = blockIdx.y * 32;
    int tx = threadIdx.x;          // 0..31
    int ty = threadIdx.y;          // 0..7
#pragma unroll
    for (int i = ty; i < 32; i += 8)
        tile[i][tx] = (_Float16)W[(size_t)(k0 + i) * N + n0 + tx];
    __syncthreads();
#pragma unroll
    for (int i = ty; i < 32; i += 8)
        Wt[(size_t)(n0 + i) * K + k0 + tx] = tile[tx][i];
}

// ---------------------------------------------------------------------------
// Gather embedding rows into an f16 activation matrix (rows = t*16+b).
// Rows >= ntok use the SOP token (decoder A_tok padding rows).
// ---------------------------------------------------------------------------
__global__ void gather_emb(const float* __restrict__ emb, const int* __restrict__ tok,
                           _Float16* __restrict__ X, int rows, int ntok, int sop)
{
    size_t gid = (size_t)blockIdx.x * blockDim.x + threadIdx.x;
    size_t total = (size_t)rows * 512;
    if (gid >= total) return;
    int row = (int)(gid >> 9);
    int e   = (int)(gid & 511);
    int t   = (row < ntok) ? tok[row] : sop;
    X[gid] = (_Float16)emb[(size_t)t * 512 + e];
}

// ---------------------------------------------------------------------------
// Persistent GRU scan, one 1024-thread workgroup (32 wave32s).
// xp: (T*16, 1536) precomputed x@Wx+b. Wht: (1536, 512) f16 transposed Wh.
// h (16x512) lives in LDS. Wave w owns output h-columns [16w, 16w+16):
// its three WMMA tiles are the r/z/n slices of hp for exactly those columns,
// so gate math is wave-local. out32/out16: (T*16, 512).
// ---------------------------------------------------------------------------
__global__ __launch_bounds__(1024)
void gru_scan(const float* __restrict__ xp, const _Float16* __restrict__ Wht,
              float* __restrict__ out32, _Float16* __restrict__ out16, int T)
{
    __shared__ float    h32[16 * 512];
    __shared__ _Float16 h16[16 * 512];

    int tid  = threadIdx.x;
    int lane = tid & 31;
    int wave = tid >> 5;

    for (int i = tid; i < 16 * 512; i += 1024) { h32[i] = 0.f; h16[i] = (_Float16)0.f; }
    __syncthreads();

    int hi  = (lane >> 4) * 8;
    int col = wave * 16 + (lane & 15);     // this lane's output h-column (C layout)
    int rb  = (lane >> 4) * 8;

    const _Float16* Ar = h16 + (size_t)(lane & 15) * 512;                      // A rows: batch
    const _Float16* Br = Wht + (size_t)(wave * 16        + (lane & 15)) * 512; // r slice
    const _Float16* Bz = Wht + (size_t)(512  + wave * 16 + (lane & 15)) * 512; // z slice
    const _Float16* Bn = Wht + (size_t)(1024 + wave * 16 + (lane & 15)) * 512; // n slice

    for (int t = 0; t < T; ++t) {
        v8f ar = {}, az = {}, an = {};
        for (int k0 = 0; k0 < 512; k0 += 32) {
            v16h a  = ldfrag(Ar + k0 + hi);
            v16h br = ldfrag(Br + k0 + hi);
            ar = __builtin_amdgcn_wmma_f32_16x16x32_f16(false, a, false, br, (short)0, ar, false, false);
            v16h bz = ldfrag(Bz + k0 + hi);
            az = __builtin_amdgcn_wmma_f32_16x16x32_f16(false, a, false, bz, (short)0, az, false, false);
            v16h bn = ldfrag(Bn + k0 + hi);
            an = __builtin_amdgcn_wmma_f32_16x16x32_f16(false, a, false, bn, (short)0, an, false, false);
        }
        float hnew[8];
#pragma unroll
        for (int r = 0; r < 8; ++r) {
            int row = rb + r;                       // batch index
            size_t xb = (size_t)(t * 16 + row) * 1536;
            float xr = xp[xb + col];
            float xz = xp[xb + 512 + col];
            float xn = xp[xb + 1024 + col];
            float rg = 1.f / (1.f + expf(-(xr + ar[r])));
            float zg = 1.f / (1.f + expf(-(xz + az[r])));
            float ng = tanhf(xn + rg * an[r]);
            float ho = h32[row * 512 + col];
            hnew[r] = (1.f - zg) * ng + zg * ho;
        }
        __syncthreads();   // all waves done reading h
#pragma unroll
        for (int r = 0; r < 8; ++r) {
            int row = rb + r;
            float hv = hnew[r];
            h32[row * 512 + col] = hv;
            h16[row * 512 + col] = (_Float16)hv;
            out32[(size_t)(t * 16 + row) * 512 + col] = hv;
            out16[(size_t)(t * 16 + row) * 512 + col] = (_Float16)hv;
        }
        __syncthreads();   // h published for next step
    }
}

// ---------------------------------------------------------------------------
// energy[p,s,b] = sum_k attn_v[k] * tanh(enc_proj[s,b,k] + q_proj[p,b,k])
// One wave per (p,s,b) triple; 8 triples per 256-thread block.
// ---------------------------------------------------------------------------
__global__ __launch_bounds__(256)
void energy_kernel(const float* __restrict__ ep, const float* __restrict__ qp,
                   const float* __restrict__ av, float* __restrict__ energy)
{
    int lane = threadIdx.x & 31;
    int wave = threadIdx.x >> 5;
    int idx  = blockIdx.x * 8 + wave;        // 0 .. 39*40*16-1
    int p = idx / 640;
    int rem = idx % 640;
    int s = rem >> 4;
    int b = rem & 15;
    const float* eprow = ep + (size_t)(s * 16 + b) * 512;
    const float* qprow = qp + (size_t)(p * 16 + b) * 512;
    float sum = 0.f;
    for (int k = lane; k < 512; k += 32)
        sum += av[k] * tanhf(eprow[k] + qprow[k]);
#pragma unroll
    for (int off = 16; off; off >>= 1) sum += __shfl_down(sum, off, 32);
    if (lane == 0) energy[((size_t)p * 40 + s) * 16 + b] = sum;
}

// softmax over s (40) per (p,b)
__global__ void softmax_kernel(const float* __restrict__ energy, float* __restrict__ attn)
{
    int gid = blockIdx.x * blockDim.x + threadIdx.x;
    if (gid >= 624) return;
    int p = gid >> 4, b = gid & 15;
    float m = -INFINITY;
    for (int s = 0; s < 40; ++s) m = fmaxf(m, energy[((size_t)p * 40 + s) * 16 + b]);
    float sum = 0.f;
    for (int s = 0; s < 40; ++s) sum += expf(energy[((size_t)p * 40 + s) * 16 + b] - m);
    float inv = 1.f / sum;
    for (int s = 0; s < 40; ++s)
        attn[((size_t)p * 40 + s) * 16 + b] = expf(energy[((size_t)p * 40 + s) * 16 + b] - m) * inv;
}

// weighted[p,b,h] = sum_s attn[p,s,b] * enc_out[s,b,h]
__global__ void weighted_kernel(const float* __restrict__ attn, const float* __restrict__ enc,
                                float* __restrict__ w32, _Float16* __restrict__ w16)
{
    int gid = blockIdx.x * 256 + threadIdx.x;   // < 624*512
    int pb = gid >> 9;
    int h  = gid & 511;
    int p  = pb >> 4;
    int b  = pb & 15;
    float s = 0.f;
    for (int si = 0; si < 40; ++si)
        s += attn[((size_t)p * 40 + si) * 16 + b] * enc[(size_t)(si * 16 + b) * 512 + h];
    w32[gid] = s;
    w16[gid] = (_Float16)s;
}

// ---------------------------------------------------------------------------
// logZ for all 7 j's of one (p,b) per block: reads the Lw row once and the 7
// Ltok rows, maintaining 7 online-LSE accumulators (saves ~43% L2 traffic vs
// one block per (p,j,b)). Output logZ[(p*7+j)*16+b].
// ---------------------------------------------------------------------------
__global__ __launch_bounds__(256)
void logz_kernel(const _Float16* __restrict__ Lt, const _Float16* __restrict__ Lw,
                 float* __restrict__ logZ)
{
    int bid = blockIdx.x;               // p*16 + b
    int p = bid >> 4;
    int b = bid & 15;

    size_t eb[7];
    eb[0] = (size_t)(640 + b) * 32000;  // SOP row
#pragma unroll
    for (int j = 1; j < 7; ++j)
        eb[j] = (size_t)(imin(p + j, 39) * 16 + b) * 32000;
    size_t wb = (size_t)(p * 16 + b) * 32000;

    float m[7], s[7];
#pragma unroll
    for (int j = 0; j < 7; ++j) { m[j] = -INFINITY; s[j] = 0.f; }

    for (int v = threadIdx.x; v < 32000; v += 256) {
        float lw = (float)Lw[wb + v];
#pragma unroll
        for (int j = 0; j < 7; ++j) {
            float x = (float)Lt[eb[j] + v] + lw;
            if (x > m[j]) { s[j] = s[j] * expf(m[j] - x) + 1.f; m[j] = x; }
            else          { s[j] += expf(x - m[j]); }
        }
    }

    __shared__ float sm[7][8], ss[7][8];
    int lane = threadIdx.x & 31, wv = threadIdx.x >> 5;
#pragma unroll
    for (int j = 0; j < 7; ++j) {
        float mj = m[j], sj = s[j];
#pragma unroll
        for (int off = 16; off; off >>= 1) {
            float mo = __shfl_down(mj, off, 32);
            float so = __shfl_down(sj, off, 32);
            float M = fmaxf(mj, mo);
            sj = sj * expf(mj - M) + so * expf(mo - M);
            mj = M;
        }
        if (lane == 0) { sm[j][wv] = mj; ss[j][wv] = sj; }
    }
    __syncthreads();
    if (threadIdx.x < 7) {
        int j = threadIdx.x;
        float M = sm[j][0], S = ss[j][0];
        for (int i = 1; i < 8; ++i) {
            float MM = fmaxf(M, sm[j][i]);
            S = S * expf(M - MM) + ss[j][i] * expf(sm[j][i] - MM);
            M = MM;
        }
        logZ[(p * 7 + j) * 16 + b] = M + logf(S);
    }
}

// Pm[p,d,b] = cumsum_d(chain_lp) + eop_lp ; one thread per (p,b)
__global__ void pm_kernel(const _Float16* __restrict__ Lt, const _Float16* __restrict__ Lw,
                          const float* __restrict__ logZ, const int* __restrict__ trg,
                          float* __restrict__ Pm)
{
    int gid = blockIdx.x * blockDim.x + threadIdx.x;
    if (gid >= 624) return;
    int p = gid >> 4, b = gid & 15;
    size_t wb = (size_t)(p * 16 + b) * 32000;
    float cum = 0.f;
    for (int d = 0; d < 6; ++d) {
        int rowC = (d == 0) ? (640 + b) : (imin(p + d, 39) * 16 + b);
        int tok  = trg[imin(p + d + 1, 39) * 16 + b];
        float lc = (float)Lt[(size_t)rowC * 32000 + tok] + (float)Lw[wb + tok]
                   - logZ[(p * 7 + d) * 16 + b];
        cum += lc;
        int je = d + 1;
        int rowE = imin(p + je, 39) * 16 + b;
        float le = (float)Lt[(size_t)rowE * 32000 + EOP_TOK] + (float)Lw[wb + EOP_TOK]
                   - logZ[(p * 7 + je) * 16 + b];
        Pm[((size_t)p * 6 + d) * 16 + b] = cum + le;
    }
}

// semi-Markov alpha recursion; one thread per batch element
__global__ void alpha_kernel(const float* __restrict__ Pm, float* __restrict__ out)
{
    int b = threadIdx.x;
    if (b >= 16) return;
    float al[40];
    al[0] = 0.f;
    for (int e = 1; e < 40; ++e) {
        float vals[6];
        float m = -INFINITY;
#pragma unroll
        for (int d = 0; d < 6; ++d) {
            int p = e - 1 - d;
            vals[d] = (p >= 0) ? (al[p] + Pm[((size_t)p * 6 + d) * 16 + b]) : -INFINITY;
            m = fmaxf(m, vals[d]);
        }
        float s = 0.f;
#pragma unroll
        for (int d = 0; d < 6; ++d) s += expf(vals[d] - m);
        al[e] = m + logf(s);
    }
    out[b] = al[39];
}

// ---------------------------------------------------------------------------
extern "C" void kernel_launch(void* const* d_in, const int* in_sizes, int n_in,
                              void* d_out, int out_size, void* d_ws, size_t ws_size,
                              hipStream_t stream)
{
    const int*   src     = (const int*)  d_in[0];
    const int*   trg     = (const int*)  d_in[1];
    const float* emb_src = (const float*)d_in[2];
    const float* emb_trg = (const float*)d_in[3];
    const float* enc_Wx  = (const float*)d_in[4];
    const float* enc_Wh  = (const float*)d_in[5];
    const float* enc_b   = (const float*)d_in[6];
    const float* tgt_Wx  = (const float*)d_in[7];
    const float* tgt_Wh  = (const float*)d_in[8];
    const float* tgt_b   = (const float*)d_in[9];
    const float* attn_We = (const float*)d_in[10];
    const float* attn_Wq = (const float*)d_in[11];
    const float* attn_v  = (const float*)d_in[12];
    const float* dec_W   = (const float*)d_in[13];
    const float* dec_b   = (const float*)d_in[14];
    float* out = (float*)d_out;

    size_t off = 0;
    auto alloc = [&](size_t bytes) -> void* {
        void* p = (char*)d_ws + off;
        off += (bytes + 255) & ~(size_t)255;
        return p;
    };
    _Float16* encWxt  = (_Float16*)alloc(1536ull * 512 * 2);
    _Float16* encWht  = (_Float16*)alloc(1536ull * 512 * 2);
    _Float16* tgtWxt  = (_Float16*)alloc(1536ull * 512 * 2);
    _Float16* tgtWht  = (_Float16*)alloc(1536ull * 512 * 2);
    _Float16* attnWet = (_Float16*)alloc(512ull * 512 * 2);
    _Float16* attnWqt = (_Float16*)alloc(512ull * 512 * 2);
    _Float16* decWt   = (_Float16*)alloc(32000ull * 1024 * 2);
    _Float16* Xsrc    = (_Float16*)alloc(640ull * 512 * 2);
    _Float16* Xtgt    = (_Float16*)alloc(640ull * 512 * 2);
    _Float16* Atok    = (_Float16*)alloc(656ull * 512 * 2);
    float*    xp_src  = (float*)   alloc(640ull * 1536 * 4);
    float*    xp_tgt  = (float*)   alloc(640ull * 1536 * 4);
    float*    enc32   = (float*)   alloc(640ull * 512 * 4);
    _Float16* enc16   = (_Float16*)alloc(640ull * 512 * 2);
    float*    tgt32   = (float*)   alloc(640ull * 512 * 4);
    _Float16* tgt16   = (_Float16*)alloc(640ull * 512 * 2);
    float*    eproj   = (float*)   alloc(640ull * 512 * 4);
    float*    qproj   = (float*)   alloc(624ull * 512 * 4);
    float*    energyB = (float*)   alloc(24960ull * 4);
    float*    attnw   = (float*)   alloc(24960ull * 4);
    float*    w32     = (float*)   alloc(624ull * 512 * 4);
    _Float16* w16     = (_Float16*)alloc(624ull * 512 * 2);
    _Float16* Ltok    = (_Float16*)alloc(656ull * 32000 * 2);
    _Float16* Lww     = (_Float16*)alloc(624ull * 32000 * 2);
    float*    logZ    = (float*)   alloc(4368ull * 4);
    float*    Pm      = (float*)   alloc(3744ull * 4);
    (void)ws_size; (void)in_sizes; (void)n_in; (void)out_size;

    // --- weight convert + transpose (f32 KxN -> f16 NxK), LDS-tiled ---
    dim3 tb(32, 8);
    cvt_transpose<<<dim3(1536 / 32, 512 / 32),  tb, 0, stream>>>(enc_Wx,  encWxt,  512, 1536);
    cvt_transpose<<<dim3(1536 / 32, 512 / 32),  tb, 0, stream>>>(enc_Wh,  encWht,  512, 1536);
    cvt_transpose<<<dim3(1536 / 32, 512 / 32),  tb, 0, stream>>>(tgt_Wx,  tgtWxt,  512, 1536);
    cvt_transpose<<<dim3(1536 / 32, 512 / 32),  tb, 0, stream>>>(tgt_Wh,  tgtWht,  512, 1536);
    cvt_transpose<<<dim3(512 / 32, 512 / 32),   tb, 0, stream>>>(attn_We, attnWet, 512, 512);
    cvt_transpose<<<dim3(512 / 32, 512 / 32),   tb, 0, stream>>>(attn_Wq, attnWqt, 512, 512);
    cvt_transpose<<<dim3(32000 / 32, 1024 / 32),tb, 0, stream>>>(dec_W,   decWt,  1024, 32000);

    // --- embedding gathers (rows = t*16+b; flat trg/src index == row) ---
    gather_emb<<<1280, 256, 0, stream>>>(emb_src, src, Xsrc, 640, 640, 0);
    gather_emb<<<1280, 256, 0, stream>>>(emb_trg, trg, Xtgt, 640, 640, 0);
    gather_emb<<<1312, 256, 0, stream>>>(emb_trg, trg, Atok, 656, 640, SOP_TOK);

    // --- xp = X @ Wx + b  (M=640, N=1536, K=512) ---
    gemm_wmma_f16<<<dim3(6, 40), 128, 0, stream>>>(Xsrc, 512, encWxt, 512,
                                                   xp_src, 1536, enc_b, 512, 0);
    gemm_wmma_f16<<<dim3(6, 40), 128, 0, stream>>>(Xtgt, 512, tgtWxt, 512,
                                                   xp_tgt, 1536, tgt_b, 512, 0);

    // --- sequential GRU scans (persistent single-WG, WMMA recurrent GEMM) ---
    gru_scan<<<1, 1024, 0, stream>>>(xp_src, encWht, enc32, enc16, 40);
    gru_scan<<<1, 1024, 0, stream>>>(xp_tgt, tgtWht, tgt32, tgt16, 40);

    // --- attention projections (M=640/624, N=512, K=512) ---
    gemm_wmma_f16<<<dim3(2, 40), 128, 0, stream>>>(enc16, 512, attnWet, 512,
                                                   eproj, 512, nullptr, 512, 0);
    gemm_wmma_f16<<<dim3(2, 39), 128, 0, stream>>>(tgt16, 512, attnWqt, 512,
                                                   qproj, 512, nullptr, 512, 0);

    energy_kernel  <<<3120, 256, 0, stream>>>(eproj, qproj, attn_v, energyB);
    softmax_kernel <<<5,    128, 0, stream>>>(energyB, attnw);
    weighted_kernel<<<1248, 256, 0, stream>>>(attnw, enc32, w32, w16);

    // --- factored decoder logits, f16 output (L2-resident) ---
    // Ltok = Atok @ dec_W[:512]   (M=656, N=32000, K=512; decWt k-range [0,512))
    gemm_wmma_f16<<<dim3(125, 41), 128, 0, stream>>>(Atok, 512, decWt, 1024,
                                                     Ltok, 32000, nullptr, 512, 1);
    // Lw = weighted @ dec_W[512:] + dec_b  (k-range [512,1024) -> decWt + 512)
    gemm_wmma_f16<<<dim3(125, 39), 128, 0, stream>>>(w16, 512, decWt + 512, 1024,
                                                     Lww, 32000, dec_b, 512, 1);

    // --- fused logsumexp over V (all 7 j per block), then Pm, alpha ---
    logz_kernel <<<624, 256, 0, stream>>>(Ltok, Lww, logZ);
    pm_kernel   <<<3,   256, 0, stream>>>(Ltok, Lww, logZ, trg, Pm);
    alpha_kernel<<<1,   32,  0, stream>>>(Pm, out);
}
